// _TanhMultiHeadAttention_74938589381252
// MI455X (gfx1250) — matrix operand order repaired and checked by reading, not compiled, not run on hardware
//
#include <hip/hip_runtime.h>
#include <hip/hip_bf16.h>

typedef _Float16 v16h __attribute__((ext_vector_type(16)));
typedef _Float16 v8h  __attribute__((ext_vector_type(8)));
typedef float    v8f  __attribute__((ext_vector_type(8)));
typedef float    v4f  __attribute__((ext_vector_type(4)));

#define NB 4
#define LL 1024
#define EE 1024
#define NH 16
#define DD 64
#define ATT_SCALE 0.125f   // 1/sqrt(64)
#define F32_TINY 1.17549435e-38f

__device__ __forceinline__ v8f wmma16(v16h a, v16h b, v8f c) {
  return __builtin_amdgcn_wmma_f32_16x16x32_f16(false, a, false, b, (short)0, c, false, false);
}

__device__ __forceinline__ float fast_tanh(float s) {
#if __has_builtin(__builtin_amdgcn_tanhf)
  return __builtin_amdgcn_tanhf(s);   // v_tanh_f32 (gfx1250 TRANS op)
#else
  float e2 = __expf(2.0f * s);
  return (e2 - 1.0f) / (e2 + 1.0f);
#endif
}

// A-fragment (16x32 f16) from row-major storage.
// caller passes per-lane base = row(m0 + lane%16) + k0 + (lane/16)*8
__device__ __forceinline__ v16h frag_a(const _Float16* base) {
  v8h lo = *(const v8h*)(base);
  v8h hi = *(const v8h*)(base + 16);
  v16h r;
#pragma unroll
  for (int i = 0; i < 8; ++i) { r[i] = lo[i]; r[i + 8] = hi[i]; }
  return r;
}

// B-fragment (32x16 f16) where B[kk][n] = T[n][kk], T row-major.
// caller passes per-lane base = Trow(n0 + lane%16) + k0 + (lane/16)*8
__device__ __forceinline__ v16h frag_b(const _Float16* base) {
  v8h lo = *(const v8h*)(base);
  v8h hi = *(const v8h*)(base + 16);
  v16h r;
#pragma unroll
  for (int i = 0; i < 8; ++i) { r[i] = lo[i]; r[i + 8] = hi[i]; }
  return r;
}

// ---------- f32 -> f16 convert (8-wide, NT reads: source is read exactly once) ----------
__global__ void cvt_kernel(const float* __restrict__ s, _Float16* __restrict__ d, int n8) {
  int i = blockIdx.x * blockDim.x + threadIdx.x;
  if (i < n8) {
    v4f a = __builtin_nontemporal_load(&((const v4f*)s)[2 * i]);
    v4f b = __builtin_nontemporal_load(&((const v4f*)s)[2 * i + 1]);
    v8h o;
#pragma unroll
    for (int j = 0; j < 4; ++j) { o[j] = (_Float16)a[j]; o[j + 4] = (_Float16)b[j]; }
    ((v8h*)d)[i] = o;
  }
}

// ---------- staged GEMM: C(4096x1024) = A(4096x1024) * T^T, A/T f16 row-major ----------
// MODE 0: store f16 as [b][h][l][d]   (Q/K layout)
// MODE 1: store f16 as [b][h][d][l]   (V transposed)
// MODE 2: store f32 as [b][l][n] scaled by fb[b]  (final output, NT)
template <int MODE>
__global__ void gemm_kernel(const _Float16* __restrict__ A, const _Float16* __restrict__ T,
                            void* __restrict__ out, const float* __restrict__ fb) {
  __shared__ _Float16 As[128 * 40];
  __shared__ _Float16 Ts[128 * 40];
  const int tid = threadIdx.x;
  const int lane = tid & 31, wv = tid >> 5;
  const int nl = lane & 15, hl = lane >> 4;
  const int mBase = blockIdx.y * 128, nBase = blockIdx.x * 128;
  const int wm = (wv >> 1) * 32;   // 4 wave-rows of 32
  const int wn = (wv & 1) * 64;    // 2 wave-cols of 64

  // staging geometry (invariant): 2 chunks of 8 halves per thread per matrix
  int row0 = tid >> 2,          off0 = (tid & 3) * 8;
  int row1 = (tid + 256) >> 2,  off1 = ((tid + 256) & 3) * 8;
  const _Float16* Ap0 = &A[(size_t)(mBase + row0) * 1024 + off0];
  const _Float16* Ap1 = &A[(size_t)(mBase + row1) * 1024 + off1];
  const _Float16* Tp0 = &T[(size_t)(nBase + row0) * 1024 + off0];
  const _Float16* Tp1 = &T[(size_t)(nBase + row1) * 1024 + off1];
  _Float16* Al0 = &As[row0 * 40 + off0];
  _Float16* Al1 = &As[row1 * 40 + off1];
  _Float16* Tl0 = &Ts[row0 * 40 + off0];
  _Float16* Tl1 = &Ts[row1 * 40 + off1];

  v8f c[2][4] = {};
  // prefetch k-step 0 into registers
  v8h ra0 = *(const v8h*)(Ap0), ra1 = *(const v8h*)(Ap1);
  v8h rt0 = *(const v8h*)(Tp0), rt1 = *(const v8h*)(Tp1);

  for (int k0 = 0; k0 < 1024; k0 += 32) {
    *(v8h*)Al0 = ra0; *(v8h*)Al1 = ra1;
    *(v8h*)Tl0 = rt0; *(v8h*)Tl1 = rt1;
    __syncthreads();
    if (k0 < 992) {  // issue next k-step's global loads; consumed next iteration
      ra0 = *(const v8h*)(Ap0 + k0 + 32); ra1 = *(const v8h*)(Ap1 + k0 + 32);
      rt0 = *(const v8h*)(Tp0 + k0 + 32); rt1 = *(const v8h*)(Tp1 + k0 + 32);
    }
    v16h af[2], bf[4];
#pragma unroll
    for (int i = 0; i < 2; ++i) af[i] = frag_a(&As[(wm + i * 16 + nl) * 40 + hl * 8]);
#pragma unroll
    for (int j = 0; j < 4; ++j) bf[j] = frag_b(&Ts[(wn + j * 16 + nl) * 40 + hl * 8]);
#pragma unroll
    for (int i = 0; i < 2; ++i)
#pragma unroll
      for (int j = 0; j < 4; ++j) c[i][j] = wmma16(af[i], bf[j], c[i][j]);
    __syncthreads();
  }
#pragma unroll
  for (int i = 0; i < 2; ++i)
#pragma unroll
    for (int j = 0; j < 4; ++j)
#pragma unroll
      for (int r = 0; r < 8; ++r) {
        int m = mBase + wm + i * 16 + r + 8 * hl;
        int n = nBase + wn + j * 16 + nl;
        float val = c[i][j][r];
        int b = m >> 10, l = m & 1023;
        if (MODE == 0) {
          int h = n >> 6, d = n & 63;
          ((_Float16*)out)[((((size_t)b * 16 + h) * 1024 + l) << 6) + d] = (_Float16)val;
        } else if (MODE == 1) {
          int h = n >> 6, d = n & 63;
          ((_Float16*)out)[(((size_t)b * 16 + h) * 64 + d) * 1024 + l] = (_Float16)val;
        } else {
          __builtin_nontemporal_store(val * fb[b], &((float*)out)[(size_t)m * 1024 + n]);
        }
      }
}

// ---------- attention: PASS 0 = sum(exp); PASS 1 = write weights + P@V ----------
template <int PASS>
__global__ void attn_kernel(const _Float16* __restrict__ Qh, const _Float16* __restrict__ Kh,
                            const _Float16* __restrict__ Vt, const unsigned char* __restrict__ mask,
                            float* __restrict__ bpart, const float* __restrict__ fb,
                            float* __restrict__ weights, _Float16* __restrict__ attU) {
  __shared__ _Float16 P16[8 * 16 * 136];  // per-wave 16x128 (+pad) P tile
  __shared__ float ssum[8];
  const int tid = threadIdx.x, lane = tid & 31, wv = tid >> 5;
  const int nl = lane & 15, hl = lane >> 4;
  const int x = blockIdx.x;
  const int qt = x & 7, h = (x >> 3) & 15, b = x >> 7;
  const size_t bh = (size_t)b * 16 + h;
  const _Float16* Qb = Qh + (bh << 10) * 64;   // [l][d]
  const _Float16* Kb = Kh + (bh << 10) * 64;   // [l][d]
  const _Float16* Vb = Vt + bh * 64 * 1024;    // [d][l]
  const unsigned char* mb = mask + b * 1024;
  const int qRow0 = qt * 128 + wv * 16;

  v16h aq[2];
#pragma unroll
  for (int ks = 0; ks < 2; ++ks)
    aq[ks] = frag_a(Qb + (size_t)(qRow0 + nl) * 64 + ks * 32 + hl * 8);

  float qm[8];
#pragma unroll
  for (int r = 0; r < 8; ++r) qm[r] = (float)mb[qRow0 + 8 * hl + r];

  float f = (PASS == 1) ? fb[b] : 0.0f;
  float lsum = 0.0f;
  v8f catt[4] = {};
  _Float16* myP = &P16[wv * 16 * 136];
  float* wrow = (PASS == 1) ? (weights + (bh << 20)) : nullptr;

  for (int kt = 0; kt < 8; ++kt) {
#pragma unroll
    for (int nt = 0; nt < 8; ++nt) {
      int kcol = kt * 128 + nt * 16 + nl;
      v8f c = {};
#pragma unroll
      for (int ks = 0; ks < 2; ++ks) {
        v16h bk = frag_b(Kb + (size_t)(kt * 128 + nt * 16 + nl) * 64 + ks * 32 + hl * 8);
        c = wmma16(aq[ks], bk, c);
      }
      float km = (float)mb[kcol];
#pragma unroll
      for (int r = 0; r < 8; ++r) {
        float s = c[r] * ATT_SCALE;
        float p = __expf(fast_tanh(s)) * km * qm[r];   // m=0 valid: |tanh|<1
        if (PASS == 0) {
          lsum += p;
        } else {
          int q = qRow0 + r + 8 * hl;
          __builtin_nontemporal_store(p * f, &wrow[(size_t)q * 1024 + kcol]);  // write-once stream
          myP[(r + 8 * hl) * 136 + nt * 16 + nl] = (_Float16)p;
        }
      }
    }
    if (PASS == 1) {  // att(16x64) += P(16x128) @ V(128x64); wave-private LDS, in-order DS
#pragma unroll
      for (int ks2 = 0; ks2 < 4; ++ks2) {
        v16h ap = frag_a(&myP[(size_t)nl * 136 + ks2 * 32 + hl * 8]);
#pragma unroll
        for (int dt = 0; dt < 4; ++dt) {
          v16h bv = frag_b(Vb + (size_t)(dt * 16 + nl) * 1024 + kt * 128 + ks2 * 32 + hl * 8);
          catt[dt] = wmma16(ap, bv, catt[dt]);
        }
      }
    }
  }

  if (PASS == 0) {
#pragma unroll
    for (int off = 16; off > 0; off >>= 1) lsum += __shfl_xor(lsum, off, 32);
    if (lane == 0) ssum[wv] = lsum;
    __syncthreads();
    if (tid == 0) {
      float t = 0.0f;
#pragma unroll
      for (int i = 0; i < 8; ++i) t += ssum[i];
      bpart[blockIdx.x] = t;   // deterministic per-block partial
    }
  } else {
#pragma unroll
    for (int dt = 0; dt < 4; ++dt)
#pragma unroll
      for (int r = 0; r < 8; ++r) {
        int q = qRow0 + r + 8 * hl;
        int e = h * 64 + dt * 16 + nl;
        attU[((size_t)b * 1024 + q) * 1024 + e] = (_Float16)catt[dt][r];
      }
  }
}

__global__ void finalize_kernel(const float* __restrict__ bpart, float* __restrict__ fb) {
  int b = threadIdx.x;
  if (b < NB) {
    float s = 0.0f;
    for (int i = 0; i < 128; ++i) s += bpart[b * 128 + i];  // blocks for batch b are contiguous
    s = fmaxf(s, F32_TINY);
    fb[b] = 1.0f / s;   // renorm == 1 analytically
  }
}

extern "C" void kernel_launch(void* const* d_in, const int* in_sizes, int n_in,
                              void* d_out, int out_size, void* d_ws, size_t ws_size,
                              hipStream_t stream) {
  (void)in_sizes; (void)n_in; (void)out_size; (void)ws_size;
  const float* x = (const float*)d_in[0];
  const unsigned char* mask = (const unsigned char*)d_in[1];
  const float* Wq = (const float*)d_in[2];
  const float* Wk = (const float*)d_in[3];
  const float* Wv = (const float*)d_in[4];
  const float* Wo = (const float*)d_in[5];

  char* ws = (char*)d_ws;
  _Float16* Xh  = (_Float16*)(ws);                         // 8 MB
  _Float16* WqH = (_Float16*)(ws + ((size_t)8  << 20));    // 2 MB
  _Float16* WkH = (_Float16*)(ws + ((size_t)10 << 20));
  _Float16* WvH = (_Float16*)(ws + ((size_t)12 << 20));
  _Float16* WoH = (_Float16*)(ws + ((size_t)14 << 20));
  _Float16* Qh  = (_Float16*)(ws + ((size_t)16 << 20));    // 8 MB [b][h][l][d]
  _Float16* Kh  = (_Float16*)(ws + ((size_t)24 << 20));    // 8 MB
  _Float16* Vt  = (_Float16*)(ws + ((size_t)32 << 20));    // 8 MB [b][h][d][l]
  _Float16* attU= (_Float16*)(ws + ((size_t)40 << 20));    // 8 MB [b][l][e]
  float* bpart  = (float*)(ws + ((size_t)48 << 20));       // 512 floats
  float* fb     = bpart + 512;                             // 4 floats

  float* outp = (float*)d_out;                              // [b][l][e] f32
  float* weights = outp + (size_t)NB * LL * EE;             // [b][h][q][k] f32

  // f32 -> f16 converts
  cvt_kernel<<<(4096 * 1024 / 8 + 255) / 256, 256, 0, stream>>>(x, Xh, 4096 * 1024 / 8);
  cvt_kernel<<<(1024 * 1024 / 8 + 255) / 256, 256, 0, stream>>>(Wq, WqH, 1024 * 1024 / 8);
  cvt_kernel<<<(1024 * 1024 / 8 + 255) / 256, 256, 0, stream>>>(Wk, WkH, 1024 * 1024 / 8);
  cvt_kernel<<<(1024 * 1024 / 8 + 255) / 256, 256, 0, stream>>>(Wv, WvH, 1024 * 1024 / 8);
  cvt_kernel<<<(1024 * 1024 / 8 + 255) / 256, 256, 0, stream>>>(Wo, WoH, 1024 * 1024 / 8);

  // Q, K, V projections (X @ W^T)
  dim3 gg(1024 / 128, 4096 / 128);
  gemm_kernel<0><<<gg, 256, 0, stream>>>(Xh, WqH, Qh, nullptr);
  gemm_kernel<0><<<gg, 256, 0, stream>>>(Xh, WkH, Kh, nullptr);
  gemm_kernel<1><<<gg, 256, 0, stream>>>(Xh, WvH, Vt, nullptr);

  // pass 0: global sum of exp(tanh(QK^T*scale)) per batch
  attn_kernel<0><<<NB * NH * 8, 256, 0, stream>>>(Qh, Kh, Vt, mask, bpart, fb, weights, attU);
  finalize_kernel<<<1, 32, 0, stream>>>(bpart, fb);

  // pass 1: recompute, write normalized weights once, fuse P@V
  attn_kernel<1><<<NB * NH * 8, 256, 0, stream>>>(Qh, Kh, Vt, mask, bpart, fb, weights, attU);

  // output = (attU @ Wo^T) * fb[b]
  gemm_kernel<2><<<gg, 256, 0, stream>>>(attU, WoH, outp, fb);
}
